// Equivariant_module_21638045237878
// MI455X (gfx1250) — compile-verified, matrix operand order checked by reading, not run on hardware
//
#include <hip/hip_runtime.h>

typedef __attribute__((ext_vector_type(16))) __bf16       v16bf;
typedef __attribute__((ext_vector_type(8)))  float        v8f;
typedef __attribute__((ext_vector_type(4)))  unsigned int v4u;
typedef __attribute__((ext_vector_type(8)))  int          v8i;
typedef __attribute__((ext_vector_type(4)))  int          v4i;

#define N_NODES 50000
#define E_EDGES 800000

// Fast sigmoid/silu: v_exp_f32 + v_add + v_rcp_f32 (avoids the IEEE div expansion;
// bf16 WMMA pipeline precision dominates anyway).
__device__ __forceinline__ float sigmoidf_(float x) {
  return __builtin_amdgcn_rcpf(1.0f + __expf(-x));
}
__device__ __forceinline__ float siluf_(float x) { return x * sigmoidf_(x); }

__device__ __forceinline__ unsigned lds_off(const void* p) {
  // low 32 bits of a generic pointer into LDS == LDS byte offset
  return (unsigned)(unsigned long long)p;
}

// Tensor Data Mover: stage a dense 2-D bf16 matrix [rows][row_elems] from global to LDS.
__device__ __forceinline__ void tdm_load_2d_bf16(const void* gptr, unsigned lds_byte_off,
                                                 unsigned rows, unsigned row_elems) {
  unsigned long long ga = (unsigned long long)gptr;
  v4u g0;
  g0[0] = 1u;                                             // count=1, user mode
  g0[1] = lds_byte_off;                                   // lds_addr
  g0[2] = (unsigned)(ga & 0xffffffffu);                   // global_addr[31:0]
  g0[3] = (unsigned)((ga >> 32) & 0x1ffffffu) | (2u << 30); // global_addr[56:32] | type=2
  v8i g1;
  g1[0] = (int)(1u << 16);                                // data_size=1 (2 bytes/elem)
  g1[1] = (int)(row_elems << 16);                         // tensor_dim0[15:0] @ [63:48]
  g1[2] = (int)(rows << 16);                              // tensor_dim1[15:0] @ [95:80]
  g1[3] = (int)(row_elems << 16);                         // tile_dim0 @ [127:112]
  g1[4] = (int)rows;                                      // tile_dim1 @ [143:128]
  g1[5] = (int)row_elems;                                 // tensor_dim0_stride[31:0]
  g1[6] = 0;
  g1[7] = 0;
  v4i z4 = {0, 0, 0, 0};
#if defined(__clang_major__) && (__clang_major__ >= 23)
  v8i z8 = {0, 0, 0, 0, 0, 0, 0, 0};
  __builtin_amdgcn_tensor_load_to_lds(g0, g1, z4, z4, z8, 0);
#else
  __builtin_amdgcn_tensor_load_to_lds(g0, g1, z4, z4, 0);
#endif
}

// Load one 16x32 bf16 A/B fragment from LDS stored row-major [row][K], K contiguous.
// ISA 16-bit A 16x32 layout: lanes 0-15 -> rows M=0..15 (K base 0), lanes 16-31 -> K base 8;
// VGPR v=0..3 hold K pairs {kb+2v, kb+2v+1}; v=4..7 hold {16+kb+2v', ...}.
__device__ __forceinline__ v16bf load_frag_bf16(const __bf16* base, int row_stride, int lane) {
  int row = lane & 15;
  int kh  = (lane >> 4) << 3;   // 0 or 8
  const __bf16* p = base + row * row_stride + kh;
  v16bf f;
#pragma unroll
  for (int v = 0; v < 4; ++v) {
    f[2 * v]         = p[2 * v];
    f[2 * v + 1]     = p[2 * v + 1];
    f[8 + 2 * v]     = p[16 + 2 * v];
    f[8 + 2 * v + 1] = p[16 + 2 * v + 1];
  }
  return f;
}

// ---------------------------------------------------------------------------
// Per-layer weight prep: bf16, transposed ("[N][K]"), K zero-padded.
// Blobs: w1T[128][32], w2T[128][128], w3T[224][128], bsT[96][416], bvT[32][288]
// ---------------------------------------------------------------------------
__global__ void prep_weights_kernel(const float* __restrict__ r0, const float* __restrict__ r1,
                                    const float* __restrict__ r2, const float* __restrict__ lin_s,
                                    const float* __restrict__ lin_v, const float* __restrict__ sc_s,
                                    const float* __restrict__ sc_v, int VIN,
                                    __bf16* __restrict__ w1T, __bf16* __restrict__ w2T,
                                    __bf16* __restrict__ w3T, __bf16* __restrict__ bsT,
                                    __bf16* __restrict__ bvT) {
  const int NW = 128 + 3 * VIN, SMSG = 64 + VIN, VMSG = 64 + 2 * VIN;
  int idx = blockIdx.x * blockDim.x + threadIdx.x;
  if (idx < 128 * 32) {            // w1T: radial0 is 20x128
    int n = idx >> 5, k = idx & 31;
    w1T[idx] = (__bf16)((k < 20) ? r0[k * 128 + n] : 0.0f);
    return;
  }
  idx -= 128 * 32;
  if (idx < 128 * 128) {           // w2T: radial1 is 128x128
    int n = idx >> 7, k = idx & 127;
    w2T[idx] = (__bf16)r1[k * 128 + n];
    return;
  }
  idx -= 128 * 128;
  if (idx < 224 * 128) {           // w3T: radial2 is 128xNW, pad rows NW..224
    int n = idx >> 7, k = idx & 127;
    w3T[idx] = (__bf16)((n < NW) ? r2[k * NW + n] : 0.0f);
    return;
  }
  idx -= 224 * 128;
  if (idx < 96 * 416) {            // bsT: [lin_s^T | sc_s^T], K = SMSG + 320 (pad to 416)
    int o = idx / 416, k = idx % 416;
    float v = 0.0f;
    if (k < SMSG) v = lin_s[k * 96 + o];
    else if (k < SMSG + 320) { int j = k - SMSG; v = sc_s[j * 96 + o]; }
    bsT[idx] = (__bf16)v;
    return;
  }
  idx -= 96 * 416;
  if (idx < 32 * 288) {            // bvT: [lin_v^T | sc_v^T], K = VMSG + VIN*5 (pad to 288)
    int o = idx / 288, k = idx % 288;
    float v = 0.0f;
    if (k < VMSG) v = lin_v[k * 32 + o];
    else if (k < VMSG + VIN * 5) { int j = k - VMSG; v = sc_v[j * 32 + o]; }
    bvT[idx] = (__bf16)v;
  }
}

__global__ void zero_kernel(float* __restrict__ p, long n) {
  for (long i = blockIdx.x * (long)blockDim.x + threadIdx.x; i < n;
       i += (long)gridDim.x * blockDim.x)
    p[i] = 0.0f;
}

// ---------------------------------------------------------------------------
// Fused radial MLP (3 WMMA GEMMs) + message build + atomic scatter.
// 64 edges / block, 128 threads (4 waves). VIN = 0 (layer 0) or 32.
// ---------------------------------------------------------------------------
template <int VIN>
__global__ void radial_msg_kernel(const int* __restrict__ esrc, const int* __restrict__ edst,
                                  const float* __restrict__ eattr, const float* __restrict__ escal,
                                  const float* __restrict__ xs, const float* __restrict__ xv,
                                  const __bf16* __restrict__ w1T, const __bf16* __restrict__ w2T,
                                  const __bf16* __restrict__ w3T, float* __restrict__ agg_s,
                                  float* __restrict__ agg_v) {
  constexpr int NW = 128 + 3 * VIN;
  constexpr int SMSG = 64 + VIN;
  constexpr int VMSG = 64 + 2 * VIN;
  extern __shared__ char smem[];
  __bf16* esA  = (__bf16*)smem;        // [64][32]
  __bf16* lw1  = esA + 64 * 32;        // [128][32]
  __bf16* lw2  = lw1 + 128 * 32;       // [128][128]
  __bf16* lw3  = lw2 + 128 * 128;      // [224][128]
  __bf16* h1   = lw3 + 224 * 128;      // [64][128]
  __bf16* h2   = h1 + 64 * 128;        // [64][128]
  __bf16* wout = h2 + 64 * 128;        // [64][NW] (alloc 64*224)
  int*   lsrc  = (int*)(wout + 64 * 224);
  int*   ldst  = lsrc + 64;
  float* lsh0  = (float*)(ldst + 64);
  float* lsh1  = lsh0 + 64;            // [64][3]

  const int tid  = threadIdx.x;
  const int lane = tid & 31;
  const int wv   = tid >> 5;
  const long eb  = (long)blockIdx.x * 64;

  if (wv == 0) {                        // TDM-stage all three weight matrices
    tdm_load_2d_bf16(w1T, lds_off(lw1), 128, 32);
    tdm_load_2d_bf16(w2T, lds_off(lw2), 128, 128);
    tdm_load_2d_bf16(w3T, lds_off(lw3), 224, 128);
    __builtin_amdgcn_s_wait_tensorcnt(0);
  }
  for (int i = tid; i < 64 * 32; i += 128) {     // stage edge scalars, pad K 20->32
    int e = i >> 5, k = i & 31;
    esA[i] = (__bf16)((k < 20) ? escal[(eb + e) * 20 + k] : 0.0f);
  }
  for (int e = tid; e < 64; e += 128) {          // stage edge meta
    lsrc[e] = esrc[eb + e];
    ldst[e] = edst[eb + e];
    const float* ea = eattr + (eb + e) * 4;
    lsh0[e] = ea[0];
    lsh1[e * 3 + 0] = ea[1]; lsh1[e * 3 + 1] = ea[2]; lsh1[e * 3 + 2] = ea[3];
  }
  __syncthreads();

  const int n0base = lane & 15;
  const int mb     = wv * 16 + ((lane >> 4) << 3);

  // GEMM1: h1 = silu(es @ W1)   M=64 K=32 N=128
  {
    v16bf a = load_frag_bf16(esA + wv * 16 * 32, 32, lane);
#pragma unroll
    for (int nt = 0; nt < 8; ++nt) {
      v16bf b = load_frag_bf16(lw1 + nt * 16 * 32, 32, lane);
      v8f c = {};
      c = __builtin_amdgcn_wmma_f32_16x16x32_bf16(false, a, false, b, (short)0, c, false, false);
      int n0 = nt * 16 + n0base;
#pragma unroll
      for (int r = 0; r < 8; ++r) h1[(mb + r) * 128 + n0] = (__bf16)siluf_(c[r]);
    }
  }
  // GEMM2: h2 = silu(h1 @ W2)   M=64 K=128 N=128
#pragma unroll
  for (int nt = 0; nt < 8; ++nt) {
    v8f c = {};
#pragma unroll
    for (int kk = 0; kk < 4; ++kk) {
      v16bf a = load_frag_bf16(h1 + (wv * 16) * 128 + kk * 32, 128, lane);
      v16bf b = load_frag_bf16(lw2 + (nt * 16) * 128 + kk * 32, 128, lane);
      c = __builtin_amdgcn_wmma_f32_16x16x32_bf16(false, a, false, b, (short)0, c, false, false);
    }
    int n0 = nt * 16 + n0base;
#pragma unroll
    for (int r = 0; r < 8; ++r) h2[(mb + r) * 128 + n0] = (__bf16)siluf_(c[r]);
  }
  // GEMM3: w = h2 @ W3          M=64 K=128 N=NW
#pragma unroll
  for (int nt = 0; nt < NW / 16; ++nt) {
    v8f c = {};
#pragma unroll
    for (int kk = 0; kk < 4; ++kk) {
      v16bf a = load_frag_bf16(h2 + (wv * 16) * 128 + kk * 32, 128, lane);
      v16bf b = load_frag_bf16(lw3 + (nt * 16) * 128 + kk * 32, 128, lane);
      c = __builtin_amdgcn_wmma_f32_16x16x32_bf16(false, a, false, b, (short)0, c, false, false);
    }
    int n0 = nt * 16 + n0base;
#pragma unroll
    for (int r = 0; r < 8; ++r) wout[(mb + r) * NW + n0] = (__bf16)c[r];
  }
  __syncthreads();

  // Message build + scatter (agg buffers live in L2: 96MB < 192MB)
  for (int i = tid; i < 64 * SMSG; i += 128) {
    int e = i / SMSG, c = i % SMSG;
    int s = lsrc[e], d = ldst[e];
    float val;
    if (c < 64) {
      val = (float)wout[e * NW + c] * xs[(long)s * 64 + c] * lsh0[e];   // wA * xs * sh0
    } else {
      int v = c - 64;                                                   // wB * (xv . sh1)
      const float* xvp = xv + (long)s * 96 + v * 3;
      float dot = xvp[0] * lsh1[e * 3 + 0] + xvp[1] * lsh1[e * 3 + 1] + xvp[2] * lsh1[e * 3 + 2];
      val = (float)wout[e * NW + 128 + v] * dot;
    }
    atomicAdd(&agg_s[(long)d * SMSG + c], 0.25f * val);
  }
  for (int i = tid; i < 64 * VMSG * 3; i += 128) {
    int e = i / (VMSG * 3);
    int r = i % (VMSG * 3);
    int vr = r / 3, k = r % 3;
    int s = lsrc[e], d = ldst[e];
    float val;
    if (vr < 64) {                                                      // wC * xs * sh1
      val = (float)wout[e * NW + 64 + vr] * xs[(long)s * 64 + vr] * lsh1[e * 3 + k];
    } else if (vr < 64 + VIN) {                                         // wD * sh0 * xv
      int v = vr - 64;
      val = (float)wout[e * NW + 128 + VIN + v] * lsh0[e] * xv[(long)s * 96 + v * 3 + k];
    } else {                                                            // wE * cross(xv, sh1)
      int v = vr - 64 - VIN;
      const float* a = xv + (long)s * 96 + v * 3;
      const float* b = lsh1 + e * 3;
      float cr = (k == 0) ? a[1] * b[2] - a[2] * b[1]
               : (k == 1) ? a[2] * b[0] - a[0] * b[2]
                          : a[0] * b[1] - a[1] * b[0];
      val = (float)wout[e * NW + 128 + 2 * VIN + v] * cr;
    }
    atomicAdd(&agg_v[(long)d * (VMSG * 3) + vr * 3 + k], 0.25f * val);
  }
}

// ---------------------------------------------------------------------------
// Node update: out_s = [agg_s | xs⊗z] @ [lin_s|sc_s],  out_v per component,
// then silu / sigmoid gating. 64 nodes / block, 128 threads.
// ---------------------------------------------------------------------------
template <int VIN>
__global__ void node_update_kernel(const float* __restrict__ agg_s, const float* __restrict__ agg_v,
                                   const float* __restrict__ xs_in, const float* __restrict__ xv_in,
                                   const float* __restrict__ z, const __bf16* __restrict__ bsT,
                                   const __bf16* __restrict__ bvT, float* __restrict__ xs_out,
                                   float* __restrict__ xv_out) {
  constexpr int SMSG = 64 + VIN, VMSG = 64 + 2 * VIN;
  constexpr int KS = 416, KV = 288;
  extern __shared__ char smem[];
  __bf16* As = (__bf16*)smem;          // [64][416]
  __bf16* Bs = As + 64 * KS;           // [96][416]
  __bf16* Av = Bs + 96 * KS;           // [64][288]
  __bf16* Bv = Av + 64 * KV;           // [32][288]
  float*  oS = (float*)(Bv + 32 * KV); // [64][96]
  float*  oV = oS + 64 * 96;           // [64][32][3]

  const int tid = threadIdx.x, lane = tid & 31, wv = tid >> 5;
  const long nb = (long)blockIdx.x * 64;
  const int n0base = lane & 15;
  const int mbase  = wv * 16 + ((lane >> 4) << 3);

  if (wv == 0) {
    tdm_load_2d_bf16(bsT, lds_off(Bs), 96, KS);
    tdm_load_2d_bf16(bvT, lds_off(Bv), 32, KV);
    __builtin_amdgcn_s_wait_tensorcnt(0);
  }
  for (int i = tid; i < 64 * KS; i += 128) {      // A_s = [agg_s | xs ⊗ z]
    int m = i / KS, k = i % KS;
    long n = nb + m;
    float v = 0.0f;
    if (n < N_NODES) {
      if (k < SMSG) v = agg_s[n * SMSG + k];
      else if (k < SMSG + 320) { int j = k - SMSG; v = xs_in[n * 64 + j / 5] * z[n * 5 + j % 5]; }
    }
    As[i] = (__bf16)v;
  }
  __syncthreads();

#pragma unroll
  for (int nt = 0; nt < 6; ++nt) {                // out_s: N=96, K=416
    v8f c = {};
#pragma unroll
    for (int kk = 0; kk < 13; ++kk) {
      v16bf a = load_frag_bf16(As + (wv * 16) * KS + kk * 32, KS, lane);
      v16bf b = load_frag_bf16(Bs + (nt * 16) * KS + kk * 32, KS, lane);
      c = __builtin_amdgcn_wmma_f32_16x16x32_bf16(false, a, false, b, (short)0, c, false, false);
    }
    int n0 = nt * 16 + n0base;
#pragma unroll
    for (int r = 0; r < 8; ++r) oS[(mbase + r) * 96 + n0] = c[r];
  }

  for (int comp = 0; comp < 3; ++comp) {          // out_v, one spatial component at a time
    __syncthreads();
    for (int i = tid; i < 64 * KV; i += 128) {    // A_v = [agg_v[:,:,c] | xv[:,:,c] ⊗ z]
      int m = i / KV, k = i % KV;
      long n = nb + m;
      float v = 0.0f;
      if (n < N_NODES) {
        if (k < VMSG) v = agg_v[n * (VMSG * 3) + k * 3 + comp];
        else if (VIN > 0 && k < VMSG + VIN * 5) {
          int j = k - VMSG;
          v = xv_in[n * 96 + (j / 5) * 3 + comp] * z[n * 5 + j % 5];
        }
      }
      Av[i] = (__bf16)v;
    }
    __syncthreads();
#pragma unroll
    for (int nt = 0; nt < 2; ++nt) {              // N=32, K=288
      v8f c = {};
#pragma unroll
      for (int kk = 0; kk < 9; ++kk) {
        v16bf a = load_frag_bf16(Av + (wv * 16) * KV + kk * 32, KV, lane);
        v16bf b = load_frag_bf16(Bv + (nt * 16) * KV + kk * 32, KV, lane);
        c = __builtin_amdgcn_wmma_f32_16x16x32_bf16(false, a, false, b, (short)0, c, false, false);
      }
      int n0 = nt * 16 + n0base;
#pragma unroll
      for (int r = 0; r < 8; ++r) oV[(mbase + r) * 96 + n0 * 3 + comp] = c[r];
    }
  }
  __syncthreads();

  for (int i = tid; i < 64 * 64; i += 128) {      // y_s = silu(out_s[:, :64])
    int m = i >> 6, c = i & 63;
    long n = nb + m;
    if (n < N_NODES) xs_out[n * 64 + c] = siluf_(oS[m * 96 + c]);
  }
  for (int i = tid; i < 64 * 32; i += 128) {      // x_v' = out_v * sigmoid(gate)
    int m = i >> 5, o = i & 31;
    long n = nb + m;
    if (n < N_NODES) {
      float g = sigmoidf_(oS[m * 96 + 64 + o]);
      xv_out[n * 96 + o * 3 + 0] = oV[m * 96 + o * 3 + 0] * g;
      xv_out[n * 96 + o * 3 + 1] = oV[m * 96 + o * 3 + 1] * g;
      xv_out[n * 96 + o * 3 + 2] = oV[m * 96 + o * 3 + 2] * g;
    }
  }
}

__global__ void concat_kernel(const float* __restrict__ xs, const float* __restrict__ xv,
                              float* __restrict__ out) {
  long i = blockIdx.x * (long)blockDim.x + threadIdx.x;
  if (i < (long)N_NODES * 160) {
    long n = i / 160;
    int c  = (int)(i % 160);
    out[i] = (c < 64) ? xs[n * 64 + c] : xv[n * 96 + (c - 64)];
  }
}

// ---------------------------------------------------------------------------
extern "C" void kernel_launch(void* const* d_in, const int* in_sizes, int n_in,
                              void* d_out, int out_size, void* d_ws, size_t ws_size,
                              hipStream_t stream) {
  (void)in_sizes; (void)n_in; (void)out_size; (void)ws_size;
  const float* x     = (const float*)d_in[0];
  const float* z     = (const float*)d_in[1];
  const int*   esrc  = (const int*)d_in[2];
  const int*   edst  = (const int*)d_in[3];
  const float* eattr = (const float*)d_in[4];
  const float* escal = (const float*)d_in[5];
  const float* P[3][7];
  for (int l = 0; l < 3; ++l)
    for (int i = 0; i < 7; ++i) P[l][i] = (const float*)d_in[6 + l * 7 + i];
  // P[l]: 0=radial0(20x128) 1=radial1(128x128) 2=radial2(128xNW)
  //       3=lin_s(smsg x96) 4=lin_v(vmsg x32) 5=sc_s(64x5x96) 6=sc_v(vin x5x32)

  char* wsp = (char*)d_ws;
  auto take = [&](size_t bytes) -> char* {
    char* p = wsp;
    wsp += (bytes + 255) & ~(size_t)255;
    return p;
  };
  __bf16 *w1T[3], *w2T[3], *w3T[3], *bsT[3], *bvT[3];
  for (int l = 0; l < 3; ++l) {
    w1T[l] = (__bf16*)take(128 * 32 * 2);
    w2T[l] = (__bf16*)take(128 * 128 * 2);
    w3T[l] = (__bf16*)take(224 * 128 * 2);
    bsT[l] = (__bf16*)take(96 * 416 * 2);
    bvT[l] = (__bf16*)take(32 * 288 * 2);
  }
  float* aggs = (float*)take((size_t)N_NODES * 96 * 4);
  float* aggv = (float*)take((size_t)N_NODES * 384 * 4);
  float* xsb[2] = {(float*)take((size_t)N_NODES * 64 * 4), (float*)take((size_t)N_NODES * 64 * 4)};
  float* xvb[2] = {(float*)take((size_t)N_NODES * 96 * 4), (float*)take((size_t)N_NODES * 96 * 4)};

  constexpr size_t SHMEM_EDGE =
      (size_t)(64 * 32 + 128 * 32 + 128 * 128 + 224 * 128 + 64 * 128 + 64 * 128 + 64 * 224) * 2 +
      (64 + 64) * 4 + (64 + 192) * 4;                         // 165,376 B
  constexpr size_t SHMEM_NODE =
      (size_t)(64 * 416 + 96 * 416 + 64 * 288 + 32 * 288) * 2 + (64 * 96 + 64 * 96) * 4; // 237,568 B

  const int EBLK = E_EDGES / 64;                 // 12500
  const int NBLK = (N_NODES + 63) / 64;          // 782

  for (int l = 0; l < 3; ++l) {
    int vin = (l == 0) ? 0 : 32;
    prep_weights_kernel<<<384, 256, 0, stream>>>(P[l][0], P[l][1], P[l][2], P[l][3], P[l][4],
                                                 P[l][5], P[l][6], vin, w1T[l], w2T[l], w3T[l],
                                                 bsT[l], bvT[l]);
  }

  // ---- layer 0 (VIN=0) ----
  zero_kernel<<<2048, 256, 0, stream>>>(aggs, (long)N_NODES * 96);
  zero_kernel<<<2048, 256, 0, stream>>>(aggv, (long)N_NODES * 384);
  radial_msg_kernel<0><<<EBLK, 128, SHMEM_EDGE, stream>>>(esrc, edst, eattr, escal, x, nullptr,
                                                          w1T[0], w2T[0], w3T[0], aggs, aggv);
  node_update_kernel<0><<<NBLK, 128, SHMEM_NODE, stream>>>(aggs, aggv, x, nullptr, z, bsT[0],
                                                           bvT[0], xsb[0], xvb[0]);
  // ---- layer 1 (VIN=32) ----
  zero_kernel<<<2048, 256, 0, stream>>>(aggs, (long)N_NODES * 96);
  zero_kernel<<<2048, 256, 0, stream>>>(aggv, (long)N_NODES * 384);
  radial_msg_kernel<32><<<EBLK, 128, SHMEM_EDGE, stream>>>(esrc, edst, eattr, escal, xsb[0],
                                                           xvb[0], w1T[1], w2T[1], w3T[1], aggs,
                                                           aggv);
  node_update_kernel<32><<<NBLK, 128, SHMEM_NODE, stream>>>(aggs, aggv, xsb[0], xvb[0], z, bsT[1],
                                                            bvT[1], xsb[1], xvb[1]);
  // ---- layer 2 (VIN=32) ----
  zero_kernel<<<2048, 256, 0, stream>>>(aggs, (long)N_NODES * 96);
  zero_kernel<<<2048, 256, 0, stream>>>(aggv, (long)N_NODES * 384);
  radial_msg_kernel<32><<<EBLK, 128, SHMEM_EDGE, stream>>>(esrc, edst, eattr, escal, xsb[1],
                                                           xvb[1], w1T[2], w2T[2], w3T[2], aggs,
                                                           aggv);
  node_update_kernel<32><<<NBLK, 128, SHMEM_NODE, stream>>>(aggs, aggv, xsb[1], xvb[1], z, bsT[2],
                                                            bvT[2], xsb[0], xvb[0]);

  concat_kernel<<<((long)N_NODES * 160 + 255) / 256, 256, 0, stream>>>(xsb[0], xvb[0],
                                                                       (float*)d_out);
}
